// PointSIFT_33071248179395
// MI455X (gfx1250) — matrix operand order repaired and checked by reading, not compiled, not run on hardware
//
#include <hip/hip_runtime.h>
#include <hip/hip_bf16.h>

#define B_      2
#define N_      4096
#define CP_     64
#define RADIUS_ 0.1f
#define EPS_    1e-5f

typedef __attribute__((ext_vector_type(16))) _Float16 v16h;
typedef __attribute__((ext_vector_type(8)))  _Float16 v8h;
typedef __attribute__((ext_vector_type(8)))  float    v8f;

// ---------------------------------------------------------------------------
// Kernel 0: fold BN into weights, convert to f16, pad layer-1 K 134->160.
// Weight layout: W[c_out][K], K = c_in*2 + kw (== OIHW flat order).
// ---------------------------------------------------------------------------
__global__ __launch_bounds__(256)
void prep_kernel(const float* __restrict__ W1, const float* __restrict__ b1,
                 const float* __restrict__ g1, const float* __restrict__ be1,
                 const float* __restrict__ mu1, const float* __restrict__ va1,
                 const float* __restrict__ W2, const float* __restrict__ b2,
                 const float* __restrict__ g2, const float* __restrict__ be2,
                 const float* __restrict__ mu2, const float* __restrict__ va2,
                 const float* __restrict__ W3, const float* __restrict__ b3,
                 const float* __restrict__ g3, const float* __restrict__ be3,
                 const float* __restrict__ mu3, const float* __restrict__ va3,
                 const float* __restrict__ W4, const float* __restrict__ b4,
                 const float* __restrict__ g4, const float* __restrict__ be4,
                 const float* __restrict__ mu4, const float* __restrict__ va4,
                 _Float16* __restrict__ W1h, _Float16* __restrict__ W2h,
                 _Float16* __restrict__ W3h, _Float16* __restrict__ W4h,
                 float* __restrict__ biases) {
    int g = blockIdx.x * 256 + threadIdx.x;
    if (g < 128 * 160) {                           // layer 1: K=134 padded to 160
        int c = g / 160, k = g % 160;
        float s = g1[c] * rsqrtf(va1[c] + EPS_);
        W1h[g] = (_Float16)((k < 134) ? W1[c * 134 + k] * s : 0.f);
        return;
    }
    g -= 128 * 160;
    if (g < 128 * 256) {                           // layer 2: K=256
        int c = g >> 8;
        float s = g2[c] * rsqrtf(va2[c] + EPS_);
        W2h[g] = (_Float16)(W2[g] * s);
        return;
    }
    g -= 128 * 256;
    if (g < 128 * 256) {                           // layer 3: K=256
        int c = g >> 8;
        float s = g3[c] * rsqrtf(va3[c] + EPS_);
        W3h[g] = (_Float16)(W3[g] * s);
        return;
    }
    g -= 128 * 256;
    if (g < 128 * 128) {                           // layer 4: K=128
        int c = g >> 7;
        float s = g4[c] * rsqrtf(va4[c] + EPS_);
        W4h[g] = (_Float16)(W4[g] * s);
        return;
    }
    g -= 128 * 128;
    if (g < 512) {                                 // folded biases, f32
        int L = g >> 7, c = g & 127;
        const float *bb, *gg, *be, *mu, *va;
        if (L == 0)      { bb = b1; gg = g1; be = be1; mu = mu1; va = va1; }
        else if (L == 1) { bb = b2; gg = g2; be = be2; mu = mu2; va = va2; }
        else if (L == 2) { bb = b3; gg = g3; be = be3; mu = mu3; va = va3; }
        else             { bb = b4; gg = g4; be = be4; mu = mu4; va = va4; }
        float s = gg[c] * rsqrtf(va[c] + EPS_);
        biases[g] = s * (bb[c] - mu[c]) + be[c];
    }
}

// ---------------------------------------------------------------------------
// Kernel 1: select_cube. One thread per point; LDS-tiled O(N^2) sweep with
// branchless per-octant min tracking (unrolled so bd/bj stay in VGPRs).
// ---------------------------------------------------------------------------
__global__ __launch_bounds__(256)
void cube_kernel(const float* __restrict__ xyz, int* __restrict__ idx) {
    __shared__ float sx[256], sy[256], sz[256];
    const int b = blockIdx.x >> 4;                 // 16 blocks per batch
    const int i = (blockIdx.x & 15) * 256 + threadIdx.x;
    const float* xb = xyz + (size_t)b * N_ * 3;
    const float xi = xb[i * 3 + 0], yi = xb[i * 3 + 1], zi = xb[i * 3 + 2];

    float bd[8];
    int   bj[8];
#pragma unroll
    for (int o = 0; o < 8; ++o) { bd[o] = __builtin_inff(); bj[o] = i; }

    for (int tile = 0; tile < N_ / 256; ++tile) {
        __syncthreads();
        int j0 = tile * 256 + threadIdx.x;
        sx[threadIdx.x] = xb[j0 * 3 + 0];
        sy[threadIdx.x] = xb[j0 * 3 + 1];
        sz[threadIdx.x] = xb[j0 * 3 + 2];
        __syncthreads();
        for (int t = 0; t < 256; ++t) {
            float dx = sx[t] - xi, dy = sy[t] - yi, dz = sz[t] - zi;
            bool within = (fabsf(dx) < RADIUS_) & (fabsf(dy) < RADIUS_) & (fabsf(dz) < RADIUS_);
            int oct = (dx > 0.f ? 4 : 0) + (dy > 0.f ? 2 : 0) + (dz > 0.f ? 1 : 0);
            float d = dx * dx + dy * dy + dz * dz;
            int j = tile * 256 + t;
#pragma unroll
            for (int o = 0; o < 8; ++o) {
                bool upd = within && (oct == o) && (d < bd[o]);  // strict < == argmin-first
                bd[o] = upd ? d : bd[o];
                bj[o] = upd ? j : bj[o];
            }
        }
    }
    int* op = idx + ((size_t)(b * N_) + i) * 8;
#pragma unroll
    for (int o = 0; o < 8; ++o) op[o] = bj[o];
}

// ---------------------------------------------------------------------------
// WMMA fragment loaders (wave32 layouts, CDNA5 ISA §7.12.2).
// A (16x32 f16): row m = lane&15; half = lane>>4; slot v -> K = (v&7)+8*half+16*(v>>3)
//   => two b128 loads at K = kk*32 + {8*half, 16+8*half}.
// B (32x16 f16): col n = lane&15; slot v -> K = (v&7)+8*half+16*(v>>3) (same as A)
//   => two b128 loads at K = kk*32 + {8*half, 16+8*half}. Activations stored [pos][K].
// ---------------------------------------------------------------------------
__device__ __forceinline__ v16h make_v16(v8h lo, v8h hi) {
    return __builtin_shufflevector(lo, hi, 0, 1, 2, 3, 4, 5, 6, 7,
                                           8, 9, 10, 11, 12, 13, 14, 15);
}
__device__ __forceinline__ v16h load_afrag(const _Float16* W, int strideH,
                                           int m, int kk, int half) {
    const _Float16* base = W + (size_t)m * strideH + kk * 32;
    v8h lo = *(const v8h*)(base + 8 * half);
    v8h hi = *(const v8h*)(base + 16 + 8 * half);
    return make_v16(lo, hi);
}
__device__ __forceinline__ v16h load_bfrag(const _Float16* act, int strideH,
                                           int n, int kk, int half) {
    const _Float16* base = act + (size_t)n * strideH + kk * 32;
    v8h lo = *(const v8h*)(base + 8 * half);
    v8h hi = *(const v8h*)(base + 16 + 8 * half);
    return make_v16(lo, hi);
}

// ---------------------------------------------------------------------------
// Kernel 2: fused gather + 4 conv/BN/ReLU layers, all intermediates in LDS.
// 16 points per block, 8 waves; wave w owns output channels [16w, 16w+16).
// ---------------------------------------------------------------------------
__global__ __launch_bounds__(256)
void pointsift_fused_kernel(const float* __restrict__ xyz,
                            const float* __restrict__ pts,
                            const int*   __restrict__ idx,
                            const _Float16* __restrict__ W1h,
                            const _Float16* __restrict__ W2h,
                            const _Float16* __restrict__ W3h,
                            const _Float16* __restrict__ W4h,
                            const float* __restrict__ biases,
                            float* __restrict__ out) {
    __shared__ __align__(32) _Float16 actA[64][160];  // layer1 in : 64 pos x K160
    __shared__ __align__(32) _Float16 actB[32][256];  // layer2 in : 32 pos x K256
    __shared__ __align__(32) _Float16 actC[16][256];  // layer3 in : 16 pos x K256
    __shared__ __align__(32) _Float16 actD[16][128];  // layer4 in : 16 pos x K128

    const int t     = threadIdx.x;
    const int lane  = t & 31;
    const int w     = t >> 5;           // wave id == M-tile id
    const int lrow  = lane & 15;
    const int lhalf = lane >> 4;

    const int b      = blockIdx.x >> 8; // 256 point-tiles per batch
    const int base_i = (blockIdx.x & 255) * 16;

    // warm caches for this wave's weight slices (global_prefetch_b8)
    __builtin_prefetch(W1h + (size_t)(w * 16) * 160, 0, 1);
    __builtin_prefetch(W2h + (size_t)(w * 16) * 256, 0, 1);
    __builtin_prefetch(W3h + (size_t)(w * 16) * 256, 0, 1);
    __builtin_prefetch(W4h + (size_t)(w * 16) * 128, 0, 1);

    // ---- zero layer-1 K padding (K = 134..159); WMMA must not see garbage ----
    for (int z = t; z < 64 * 26; z += 256)
        actA[z / 26][134 + (z % 26)] = (_Float16)0.f;

    // ---- gather: 128 (point, neighbor) pairs, 2 threads per pair ------------
    {
        const int pn = t >> 1;          // 0..127
        const int hs = t & 1;
        const int p  = pn >> 3;
        const int nb = pn & 7;
        const int i  = base_i + p;
        const int j  = idx[((size_t)(b * N_) + i) * 8 + nb];
        const int pos = p * 4 + (nb >> 1);   // conv1_0 output position
        const int kw  = nb & 1;              // conv1_0 tap
        const float* xi = xyz + (size_t)(b * N_ + i) * 3;
        const float* xj = xyz + (size_t)(b * N_ + j) * 3;
        if (hs == 0) {
            actA[pos][0 * 2 + kw] = (_Float16)(xj[0] - xi[0]);
            actA[pos][1 * 2 + kw] = (_Float16)(xj[1] - xi[1]);
            actA[pos][2 * 2 + kw] = (_Float16)(xj[2] - xi[2]);
        }
        const float* pj = pts + (size_t)(b * N_ + j) * CP_;
        const int c0 = hs * 32;
#pragma unroll 8
        for (int c = 0; c < 32; ++c)
            actA[pos][(3 + c0 + c) * 2 + kw] = (_Float16)pj[c0 + c];
    }
    __syncthreads();

    // ---- layer 1: M=128, K=160, Npos=64 (4 N-tiles) -------------------------
    {
        float bch[8];
#pragma unroll
        for (int v = 0; v < 8; ++v) bch[v] = biases[0 * 128 + w * 16 + v + 8 * lhalf];
        for (int nt = 0; nt < 4; ++nt) {
            v8f acc = {};
#pragma unroll
            for (int kk = 0; kk < 5; ++kk) {
                v16h a  = load_afrag(W1h, 160, w * 16 + lrow, kk, lhalf);
                v16h bv = load_bfrag(&actA[0][0], 160, nt * 16 + lrow, kk, lhalf);
                acc = __builtin_amdgcn_wmma_f32_16x16x32_f16(
                    false, a, false, bv, (short)0, acc, false, false);
            }
            const int pos1  = nt * 16 + lrow;          // = p*4 + w1pos
            const int p     = pos1 >> 2;
            const int w1pos = pos1 & 3;
            const int row2  = p * 2 + (w1pos >> 1);    // layer-2 position
            const int kw2   = w1pos & 1;               // layer-2 tap
#pragma unroll
            for (int v = 0; v < 8; ++v) {
                int ch = w * 16 + v + 8 * lhalf;
                float val = acc[v] + bch[v];
                actB[row2][ch * 2 + kw2] = (_Float16)(val > 0.f ? val : 0.f);
            }
        }
    }
    __syncthreads();

    // ---- layer 2: M=128, K=256, Npos=32 (2 N-tiles) -------------------------
    {
        float bch[8];
#pragma unroll
        for (int v = 0; v < 8; ++v) bch[v] = biases[1 * 128 + w * 16 + v + 8 * lhalf];
        for (int nt = 0; nt < 2; ++nt) {
            v8f acc = {};
#pragma unroll
            for (int kk = 0; kk < 8; ++kk) {
                v16h a  = load_afrag(W2h, 256, w * 16 + lrow, kk, lhalf);
                v16h bv = load_bfrag(&actB[0][0], 256, nt * 16 + lrow, kk, lhalf);
                acc = __builtin_amdgcn_wmma_f32_16x16x32_f16(
                    false, a, false, bv, (short)0, acc, false, false);
            }
            const int pos2 = nt * 16 + lrow;           // = p*2 + w2pos
            const int p    = pos2 >> 1;
            const int kw3  = pos2 & 1;                 // layer-3 tap
#pragma unroll
            for (int v = 0; v < 8; ++v) {
                int ch = w * 16 + v + 8 * lhalf;
                float val = acc[v] + bch[v];
                actC[p][ch * 2 + kw3] = (_Float16)(val > 0.f ? val : 0.f);
            }
        }
    }
    __syncthreads();

    // ---- layer 3: M=128, K=256, Npos=16 -------------------------------------
    {
        float bch[8];
#pragma unroll
        for (int v = 0; v < 8; ++v) bch[v] = biases[2 * 128 + w * 16 + v + 8 * lhalf];
        v8f acc = {};
#pragma unroll
        for (int kk = 0; kk < 8; ++kk) {
            v16h a  = load_afrag(W3h, 256, w * 16 + lrow, kk, lhalf);
            v16h bv = load_bfrag(&actC[0][0], 256, lrow, kk, lhalf);
            acc = __builtin_amdgcn_wmma_f32_16x16x32_f16(
                false, a, false, bv, (short)0, acc, false, false);
        }
#pragma unroll
        for (int v = 0; v < 8; ++v) {
            int ch = w * 16 + v + 8 * lhalf;
            float val = acc[v] + bch[v];
            actD[lrow][ch] = (_Float16)(val > 0.f ? val : 0.f);  // kernel-width-1 next
        }
    }
    __syncthreads();

    // ---- layer 4: M=128, K=128, Npos=16; write (B,128,N) output -------------
    {
        float bch[8];
#pragma unroll
        for (int v = 0; v < 8; ++v) bch[v] = biases[3 * 128 + w * 16 + v + 8 * lhalf];
        v8f acc = {};
#pragma unroll
        for (int kk = 0; kk < 4; ++kk) {
            v16h a  = load_afrag(W4h, 128, w * 16 + lrow, kk, lhalf);
            v16h bv = load_bfrag(&actD[0][0], 128, lrow, kk, lhalf);
            acc = __builtin_amdgcn_wmma_f32_16x16x32_f16(
                false, a, false, bv, (short)0, acc, false, false);
        }
        const int i = base_i + lrow;
#pragma unroll
        for (int v = 0; v < 8; ++v) {
            int ch = w * 16 + v + 8 * lhalf;
            float val = acc[v] + bch[v];
            out[((size_t)b * 128 + ch) * N_ + i] = (val > 0.f ? val : 0.f);
        }
    }
}

// ---------------------------------------------------------------------------
extern "C" void kernel_launch(void* const* d_in, const int* in_sizes, int n_in,
                              void* d_out, int out_size, void* d_ws, size_t ws_size,
                              hipStream_t stream) {
    const float* xyz = (const float*)d_in[0];
    const float* pts = (const float*)d_in[1];
    const float* P[24];
    for (int k = 0; k < 24; ++k) P[k] = (const float*)d_in[2 + k];
    float* out = (float*)d_out;

    // workspace layout (all 32B-aligned offsets)
    char* ws = (char*)d_ws;
    int*      idx  = (int*)ws;                        //  2*4096*8 ints = 262144 B
    _Float16* W1h  = (_Float16*)(ws + 262144);        //  128*160*2 = 40960 B
    _Float16* W2h  = (_Float16*)(ws + 303104);        //  128*256*2 = 65536 B
    _Float16* W3h  = (_Float16*)(ws + 368640);        //  65536 B
    _Float16* W4h  = (_Float16*)(ws + 434176);        //  128*128*2 = 32768 B
    float*    bias = (float*)(ws + 466944);           //  4*128 floats

    // 0) BN-fold + f16 weight prep: 102912 elements
    hipLaunchKernelGGL(prep_kernel, dim3(402), dim3(256), 0, stream,
                       P[0],  P[1],  P[2],  P[3],  P[4],  P[5],
                       P[6],  P[7],  P[8],  P[9],  P[10], P[11],
                       P[12], P[13], P[14], P[15], P[16], P[17],
                       P[18], P[19], P[20], P[21], P[22], P[23],
                       W1h, W2h, W3h, W4h, bias);

    // 1) octant nearest-neighbor search
    hipLaunchKernelGGL(cube_kernel, dim3(B_ * (N_ / 256)), dim3(256), 0, stream,
                       xyz, idx);

    // 2) fused gather + 4x (conv+BN+ReLU) via WMMA; 16 points per block
    hipLaunchKernelGGL(pointsift_fused_kernel, dim3(B_ * (N_ / 16)), dim3(256), 0,
                       stream, xyz, pts, idx, W1h, W2h, W3h, W4h, bias, out);
}
